// MaskedAttention_78683800862789
// MI455X (gfx1250) — matrix operand order repaired and checked by reading, not compiled, not run on hardware
//
#include <hip/hip_runtime.h>

typedef __bf16 bf16_t;
typedef __bf16 v4bf  __attribute__((ext_vector_type(4)));
typedef __bf16 v8bf  __attribute__((ext_vector_type(8)));
typedef __bf16 v16bf __attribute__((ext_vector_type(16)));
typedef float  v8f   __attribute__((ext_vector_type(8)));
typedef unsigned int v4u __attribute__((ext_vector_type(4)));
typedef int    v4i   __attribute__((ext_vector_type(4)));
typedef int    v8i   __attribute__((ext_vector_type(8)));

#define B_SZ   2
#define SEQ    2048
#define DIM    2048
#define NH     32
#define NKV    8
#define HD     64
#define MTOT   (B_SZ * SEQ)          // 4096
#define LDSP   40                    // LDS row stride: 32 data + 8 pad bf16 (80B)

#if __has_builtin(__builtin_amdgcn_tensor_load_to_lds)
#define HAVE_TDM 1
#else
#define HAVE_TDM 0
#endif

static __device__ __forceinline__ v8f vzero8() {
  v8f v;
#pragma unroll
  for (int i = 0; i < 8; ++i) v[i] = 0.0f;
  return v;
}

static __device__ __forceinline__ v16bf cat8(v8bf lo, v8bf hi) {
  return __builtin_shufflevector(lo, hi, 0,1,2,3,4,5,6,7,8,9,10,11,12,13,14,15);
}

static __device__ __forceinline__ v8f wmma_bf16(v16bf a, v16bf b, v8f c) {
  return __builtin_amdgcn_wmma_f32_16x16x32_bf16(false, a, false, b, (short)0, c,
                                                 false, false);
}

// A-fragment (16x32, 16-bit): lane holds row (L&15); elems 0..7 = K half*8..,
// elems 8..15 = K 16+half*8.. (half = L>>4)
static __device__ __forceinline__ v16bf load_a_frag(const bf16_t* row, int half) {
  v8bf lo = *(const v8bf*)(row + half * 8);
  v8bf hi = *(const v8bf*)(row + 16 + half * 8);
  return cat8(lo, hi);
}

// B-fragment (32x16, 16-bit): lane holds column (L&15); elems 0..7 = K half*8..,
// elems 8..15 = K 16+half*8.. (B stored [n][k]).
static __device__ __forceinline__ v16bf load_b_frag(const bf16_t* colrow, int half) {
  v8bf lo = *(const v8bf*)(colrow + half * 8);
  v8bf hi = *(const v8bf*)(colrow + 16 + half * 8);
  return cat8(lo, hi);
}

// ---------------------------------------------------------------------------
// TDM: async-load a 2D tile (tile_dim0 = 32 bf16 along rows, tile_dim1 = 128
// rows) into LDS with hardware padding: 16 DWORDs data + 4 DWORDs pad per row
// -> LDS row stride = 40 bf16 = LDSP.  Descriptor per CDNA5 ISA ch.8 (D#).
// ---------------------------------------------------------------------------
static __device__ __forceinline__ void tdm_load_tile(
    const bf16_t* gsrc, unsigned lds_byte_off, unsigned tensor_rows,
    unsigned row_len_elems)
{
#if HAVE_TDM
  unsigned long long ga = (unsigned long long)(size_t)gsrc;
  v4u g0;
  g0[0] = 1u;                                   // count=1 (valid), user mode
  g0[1] = lds_byte_off;                         // lds_addr
  g0[2] = (unsigned)ga;                         // global_addr[31:0]
  g0[3] = (unsigned)(ga >> 32) | (2u << 30);    // global_addr[56:32] | type=2
  v8i g1;
  g1[0] = (int)((1u << 16)        // data_size = 1 -> 2 bytes
              | (1u << 20)        // pad_enable
              | (3u << 22)        // pad_interval: 16 DWORDs between pads
              | (3u << 25));      // pad_amount: 4 DWORDs of padding
  g1[1] = (int)((row_len_elems & 0xffffu) << 16);                 // dim0[15:0]
  g1[2] = (int)((row_len_elems >> 16) | ((tensor_rows & 0xffffu) << 16));
  g1[3] = (int)((tensor_rows >> 16) | (32u << 16));               // tile_dim0=32
  g1[4] = (int)128u;                        // tile_dim1=128, tile_dim2=0(unused)
  g1[5] = (int)row_len_elems;               // tensor_dim0_stride[31:0]
  g1[6] = 0;                                // stride0[47:32]=0, stride1 lo=0
  g1[7] = 0;
  v4i g2; g2[0] = 1; g2[1] = 1; g2[2] = 0; g2[3] = 0;   // dims2/3=1, tile_dim3=0
  v4i g3; g3[0] = 0; g3[1] = (int)(1u << 16); g3[2] = 0; g3[3] = 0; // dim4=1
#if defined(__clang_major__) && (__clang_major__ >= 23)
  v8i gx; gx[0]=0; gx[1]=0; gx[2]=0; gx[3]=0; gx[4]=0; gx[5]=0; gx[6]=0; gx[7]=0;
  __builtin_amdgcn_tensor_load_to_lds(g0, g1, g2, g3, gx, 0);
#else
  __builtin_amdgcn_tensor_load_to_lds(g0, g1, g2, g3, 0);
#endif
#else
  (void)gsrc; (void)lds_byte_off; (void)tensor_rows; (void)row_len_elems;
#endif
}

static __device__ __forceinline__ void tdm_wait() {
#if HAVE_TDM
  __builtin_amdgcn_s_wait_tensorcnt((short)0);
#endif
}

// Fallback synchronous fill (used only if TDM builtin absent)
static __device__ __forceinline__ void fill_tile_sync(
    bf16_t (*dst)[LDSP], const bf16_t* src, int t)
{
#pragma unroll
  for (int pass = 0; pass < 2; ++pass) {
    int row = (t >> 2) + pass * 64;
    int kq  = (t & 3) * 8;
    *(v8bf*)(&dst[row][kq]) = *(const v8bf*)(src + (size_t)row * DIM + kq);
  }
}

// ---------------------------------------------------------------------------
// Kernel 0a: f32 -> bf16 (vectorized x4)
// ---------------------------------------------------------------------------
__global__ __launch_bounds__(256) void cvt_bf16_kernel(
    const float* __restrict__ s, bf16_t* __restrict__ d, int n4)
{
  int i = blockIdx.x * 256 + threadIdx.x;
  if (i >= n4) return;
  float4 f = ((const float4*)s)[i];
  v4bf o;
  o[0] = (bf16_t)f.x; o[1] = (bf16_t)f.y; o[2] = (bf16_t)f.z; o[3] = (bf16_t)f.w;
  ((v4bf*)d)[i] = o;
}

// ---------------------------------------------------------------------------
// Kernel 0b: f32 [K][N] -> bf16 [N][K] (convert + transpose, 32x32 LDS tiles)
// ---------------------------------------------------------------------------
__global__ __launch_bounds__(256) void cvtT_bf16_kernel(
    const float* __restrict__ w, bf16_t* __restrict__ wt, int K, int N)
{
  __shared__ float tile[32][33];
  int bx = blockIdx.x * 32;          // n
  int by = blockIdx.y * 32;          // k
  int tx = threadIdx.x & 31, ty = threadIdx.x >> 5;
#pragma unroll
  for (int p = 0; p < 4; ++p)
    tile[ty + p * 8][tx] = w[(size_t)(by + ty + p * 8) * N + bx + tx];
  __syncthreads();
#pragma unroll
  for (int p = 0; p < 4; ++p)
    wt[(size_t)(bx + ty + p * 8) * K + by + tx] = (bf16_t)tile[tx][ty + p * 8];
}

// ---------------------------------------------------------------------------
// Kernel 1: fused QKV GEMM, TDM double-buffered LDS, bf16 WMMA, f32 accum.
// A = xb [4096][2048] bf16, B = transposed weights [N][2048] bf16.
// V written transposed ([b][kvh][d][seq]) for the attention P*V operand.
// ---------------------------------------------------------------------------
__global__ __launch_bounds__(256) void qkv_gemm_kernel(
    const bf16_t* __restrict__ xb,  const bf16_t* __restrict__ wqt,
    const bf16_t* __restrict__ wkt, const bf16_t* __restrict__ wvt,
    bf16_t* __restrict__ qb, bf16_t* __restrict__ kb, bf16_t* __restrict__ vtb)
{
  __shared__ __align__(16) bf16_t As[2][128][LDSP];
  __shared__ __align__(16) bf16_t Bs[2][128][LDSP];

  const int t    = threadIdx.x;
  const int lane = t & 31;
  const int wave = t >> 5;
  const int half = lane >> 4;
  const int ln   = lane & 15;
  const int m0   = blockIdx.x * 128;
  const int n0   = blockIdx.y * 128;

  const bf16_t* wt; int cbase, region; unsigned brows;
  if (n0 < 2048)      { wt = wqt; cbase = 0;    region = 0; brows = 2048; }
  else if (n0 < 2560) { wt = wkt; cbase = 2048; region = 1; brows = 512; }
  else                { wt = wvt; cbase = 2560; region = 2; brows = 512; }

  const bf16_t* asrc = xb + (size_t)m0 * DIM;
  const bf16_t* bsrc = wt + (size_t)(n0 - cbase) * DIM;

  const int wm = wave >> 1;
  const int wn = wave & 1;

  v8f acc[2][4];
#pragma unroll
  for (int i = 0; i < 2; ++i)
#pragma unroll
    for (int j = 0; j < 4; ++j) acc[i][j] = vzero8();

#if HAVE_TDM
  const unsigned ldsA0 = (unsigned)(size_t)&As[0][0][0];
  const unsigned ldsA1 = (unsigned)(size_t)&As[1][0][0];
  const unsigned ldsB0 = (unsigned)(size_t)&Bs[0][0][0];
  const unsigned ldsB1 = (unsigned)(size_t)&Bs[1][0][0];
  if (wave == 0) {
    tdm_load_tile(asrc, ldsA0, MTOT, DIM);
    tdm_load_tile(bsrc, ldsB0, brows, DIM);
    tdm_wait();
  }
  __syncthreads();
#endif

  for (int k0 = 0; k0 < DIM; k0 += 32) {
    const int cur = (k0 >> 5) & 1;
#if HAVE_TDM
    if (wave == 0 && (k0 + 32) < DIM) {      // async-prefetch next tile pair
      tdm_load_tile(asrc + k0 + 32, cur ? ldsA0 : ldsA1, MTOT, DIM);
      tdm_load_tile(bsrc + k0 + 32, cur ? ldsB0 : ldsB1, brows, DIM);
    }
#else
    __syncthreads();
    fill_tile_sync(As[cur], asrc + k0, t);
    fill_tile_sync(Bs[cur], bsrc + k0, t);
    __syncthreads();
#endif
    v16bf af[2], bf[4];
#pragma unroll
    for (int i = 0; i < 2; ++i)
      af[i] = load_a_frag(&As[cur][wm * 32 + i * 16 + ln][0], half);
#pragma unroll
    for (int j = 0; j < 4; ++j)
      bf[j] = load_b_frag(&Bs[cur][wn * 64 + j * 16 + ln][0], half);
#pragma unroll
    for (int i = 0; i < 2; ++i)
#pragma unroll
      for (int j = 0; j < 4; ++j)
        acc[i][j] = wmma_bf16(af[i], bf[j], acc[i][j]);
#if HAVE_TDM
    __syncthreads();                          // all waves done reading cur
    if (wave == 0) tdm_wait();                // next buffer landed
    __syncthreads();                          // publish
#endif
  }

#pragma unroll
  for (int i = 0; i < 2; ++i) {
#pragma unroll
    for (int j = 0; j < 4; ++j) {
#pragma unroll
      for (int r = 0; r < 8; ++r) {
        int gm = m0 + wm * 32 + i * 16 + half * 8 + r;
        int gc = n0 + wn * 64 + j * 16 + ln;
        float val = acc[i][j][r];
        if (region == 0) {
          qb[(size_t)gm * 2048 + gc] = (bf16_t)val;
        } else if (region == 1) {
          kb[(size_t)gm * 512 + (gc - 2048)] = (bf16_t)val;
        } else {
          int vc = gc - 2560;
          int kvh = vc >> 6, d = vc & 63;
          int bb = gm >> 11, tok = gm & 2047;
          vtb[((size_t)(bb * NKV + kvh) * HD + d) * SEQ + tok] = (bf16_t)val;
        }
      }
    }
  }
}

// ---------------------------------------------------------------------------
// Kernel 2: RoPE (in-place on bf16 Q or K)
// ---------------------------------------------------------------------------
__global__ __launch_bounds__(256) void rope_kernel(
    bf16_t* __restrict__ buf, const float* __restrict__ cs,
    const float* __restrict__ sn, int hshift, int total)
{
  int idx = blockIdx.x * 256 + threadIdx.x;
  if (idx >= total) return;
  int j  = idx & 31;
  int t2 = idx >> 5;
  int hh = t2 & ((1 << hshift) - 1);
  int m  = t2 >> hshift;
  int s  = m & (SEQ - 1);
  size_t base = (((size_t)m << hshift) + hh) * HD + 2 * j;
  float xr = (float)buf[base];
  float xi = (float)buf[base + 1];
  float c  = cs[s * 32 + j];
  float si = sn[s * 32 + j];
  buf[base]     = (bf16_t)(xr * c - xi * si);
  buf[base + 1] = (bf16_t)(xr * si + xi * c);
}

// ---------------------------------------------------------------------------
// Kernel 3: flash attention.  1 wave = 16 q-rows x head_dim 64, 32-key chunks,
// online softmax, WMMA for QK^T and P*V; K/V fragments load straight from
// L2-resident global memory.
// ---------------------------------------------------------------------------
__global__ __launch_bounds__(128) void attn_kernel(
    const bf16_t* __restrict__ q, const bf16_t* __restrict__ k,
    const bf16_t* __restrict__ vt, const float* __restrict__ mask,
    bf16_t* __restrict__ attn)
{
  __shared__ __align__(16) bf16_t P[4][16][LDSP];
  const int lane = threadIdx.x & 31;
  const int wave = threadIdx.x >> 5;
  const int half = lane >> 4;
  const int ln   = lane & 15;
  const int h    = blockIdx.y;
  const int b    = blockIdx.z;
  const int kvh  = h >> 2;
  const int q0   = blockIdx.x * 64 + wave * 16;

  const bf16_t* qbase = q + ((size_t)(b * SEQ + q0)) * 2048 + h * HD;
  v16bf qa[2];
#pragma unroll
  for (int kk = 0; kk < 2; ++kk)
    qa[kk] = load_a_frag(qbase + (size_t)ln * 2048 + kk * 32, half);

  v8f o[4];
#pragma unroll
  for (int j = 0; j < 4; ++j) o[j] = vzero8();
  float mi[8], li[8];
#pragma unroll
  for (int r = 0; r < 8; ++r) { mi[r] = -1e30f; li[r] = 0.0f; }

  const bf16_t* kbase = k  + (size_t)b * SEQ * 512 + kvh * HD;
  const bf16_t* vbase = vt + ((size_t)(b * NKV + kvh) * HD) * SEQ;

  for (int kt = 0; kt < SEQ; kt += 32) {
    if (kt + 32 < SEQ) {
      __builtin_prefetch(kbase + (size_t)(kt + 32 + ln) * 512, 0, 0);
      __builtin_prefetch(vbase + (size_t)ln * SEQ + kt + 32, 0, 0);
    }
    v8f s0 = vzero8(), s1 = vzero8();
#pragma unroll
    for (int kk = 0; kk < 2; ++kk) {
      v16bf b0 = load_b_frag(kbase + (size_t)(kt + ln) * 512 + kk * 32, half);
      s0 = wmma_bf16(qa[kk], b0, s0);
      v16bf b1 = load_b_frag(kbase + (size_t)(kt + 16 + ln) * 512 + kk * 32, half);
      s1 = wmma_bf16(qa[kk], b1, s1);
    }
    float corr[8];
#pragma unroll
    for (int r = 0; r < 8; ++r) {
      int qg = q0 + half * 8 + r;
      float e0 = s0[r] * 0.125f + mask[(size_t)qg * SEQ + kt + ln];
      float e1 = s1[r] * 0.125f + mask[(size_t)qg * SEQ + kt + 16 + ln];
      float rm = fmaxf(e0, e1);
      rm = fmaxf(rm, __shfl_xor(rm, 1, 32));
      rm = fmaxf(rm, __shfl_xor(rm, 2, 32));
      rm = fmaxf(rm, __shfl_xor(rm, 4, 32));
      rm = fmaxf(rm, __shfl_xor(rm, 8, 32));
      float mn = fmaxf(mi[r], rm);
      float p0 = __expf(e0 - mn);
      float p1 = __expf(e1 - mn);
      float rs = p0 + p1;
      rs += __shfl_xor(rs, 1, 32);
      rs += __shfl_xor(rs, 2, 32);
      rs += __shfl_xor(rs, 4, 32);
      rs += __shfl_xor(rs, 8, 32);
      corr[r] = __expf(mi[r] - mn);
      li[r] = li[r] * corr[r] + rs;
      mi[r] = mn;
      P[wave][half * 8 + r][ln]      = (bf16_t)p0;   // acc-layout -> LDS
      P[wave][half * 8 + r][16 + ln] = (bf16_t)p1;
    }
#pragma unroll
    for (int j = 0; j < 4; ++j)
#pragma unroll
      for (int r = 0; r < 8; ++r) o[j][r] *= corr[r];

    v16bf pa = load_a_frag(&P[wave][ln][0], half);   // same-wave LDS: in order
#pragma unroll
    for (int j = 0; j < 4; ++j) {
      v16bf vb = load_b_frag(vbase + (size_t)(j * 16 + ln) * SEQ + kt, half);
      o[j] = wmma_bf16(pa, vb, o[j]);
    }
  }

#pragma unroll
  for (int j = 0; j < 4; ++j)
#pragma unroll
    for (int r = 0; r < 8; ++r) {
      int qg = q0 + half * 8 + r;
      attn[((size_t)(b * SEQ + qg)) * 2048 + h * HD + j * 16 + ln] =
          (bf16_t)(o[j][r] / li[r]);
    }
}

// ---------------------------------------------------------------------------
// Kernel 4: output projection (TDM double-buffered), f32 output.
// ---------------------------------------------------------------------------
__global__ __launch_bounds__(256) void out_gemm_kernel(
    const bf16_t* __restrict__ a, const bf16_t* __restrict__ wot,
    float* __restrict__ out)
{
  __shared__ __align__(16) bf16_t As[2][128][LDSP];
  __shared__ __align__(16) bf16_t Bs[2][128][LDSP];

  const int t    = threadIdx.x;
  const int lane = t & 31;
  const int wave = t >> 5;
  const int half = lane >> 4;
  const int ln   = lane & 15;
  const int m0   = blockIdx.x * 128;
  const int n0   = blockIdx.y * 128;
  const int wm   = wave >> 1;
  const int wn   = wave & 1;

  const bf16_t* asrc = a   + (size_t)m0 * DIM;
  const bf16_t* bsrc = wot + (size_t)n0 * DIM;

  v8f acc[2][4];
#pragma unroll
  for (int i = 0; i < 2; ++i)
#pragma unroll
    for (int j = 0; j < 4; ++j) acc[i][j] = vzero8();

#if HAVE_TDM
  const unsigned ldsA0 = (unsigned)(size_t)&As[0][0][0];
  const unsigned ldsA1 = (unsigned)(size_t)&As[1][0][0];
  const unsigned ldsB0 = (unsigned)(size_t)&Bs[0][0][0];
  const unsigned ldsB1 = (unsigned)(size_t)&Bs[1][0][0];
  if (wave == 0) {
    tdm_load_tile(asrc, ldsA0, MTOT, DIM);
    tdm_load_tile(bsrc, ldsB0, DIM,  DIM);
    tdm_wait();
  }
  __syncthreads();
#endif

  for (int k0 = 0; k0 < DIM; k0 += 32) {
    const int cur = (k0 >> 5) & 1;
#if HAVE_TDM
    if (wave == 0 && (k0 + 32) < DIM) {
      tdm_load_tile(asrc + k0 + 32, cur ? ldsA0 : ldsA1, MTOT, DIM);
      tdm_load_tile(bsrc + k0 + 32, cur ? ldsB0 : ldsB1, DIM,  DIM);
    }
#else
    __syncthreads();
    fill_tile_sync(As[cur], asrc + k0, t);
    fill_tile_sync(Bs[cur], bsrc + k0, t);
    __syncthreads();
#endif
    v16bf af[2], bf[4];
#pragma unroll
    for (int i = 0; i < 2; ++i)
      af[i] = load_a_frag(&As[cur][wm * 32 + i * 16 + ln][0], half);
#pragma unroll
    for (int j = 0; j < 4; ++j)
      bf[j] = load_b_frag(&Bs[cur][wn * 64 + j * 16 + ln][0], half);
#pragma unroll
    for (int i = 0; i < 2; ++i)
#pragma unroll
      for (int j = 0; j < 4; ++j)
        acc[i][j] = wmma_bf16(af[i], bf[j], acc[i][j]);
#if HAVE_TDM
    __syncthreads();
    if (wave == 0) tdm_wait();
    __syncthreads();
#endif
  }

#pragma unroll
  for (int i = 0; i < 2; ++i)
#pragma unroll
    for (int j = 0; j < 4; ++j)
#pragma unroll
      for (int r = 0; r < 8; ++r) {
        int gm = m0 + wm * 32 + i * 16 + half * 8 + r;
        int gc = n0 + wn * 64 + j * 16 + ln;
        out[(size_t)gm * DIM + gc] = acc[i][j][r];
      }
}

// ---------------------------------------------------------------------------
extern "C" void kernel_launch(void* const* d_in, const int* in_sizes, int n_in,
                              void* d_out, int out_size, void* d_ws, size_t ws_size,
                              hipStream_t stream) {
  (void)in_sizes; (void)n_in; (void)out_size; (void)ws_size;
  const float* x    = (const float*)d_in[0];
  const float* fc   = (const float*)d_in[1];
  const float* fs   = (const float*)d_in[2];
  const float* mask = (const float*)d_in[3];
  const float* wq   = (const float*)d_in[4];
  const float* wk   = (const float*)d_in[5];
  const float* wv   = (const float*)d_in[6];
  const float* wo   = (const float*)d_in[7];
  float* out = (float*)d_out;

  // Workspace (bf16): xb | wqt | wkt | wvt | wot | Q | K | V^T | attn (~80 MB)
  bf16_t* xb   = (bf16_t*)d_ws;
  bf16_t* wqt  = xb   + (size_t)MTOT * DIM;
  bf16_t* wkt  = wqt  + (size_t)DIM * DIM;
  bf16_t* wvt  = wkt  + (size_t)512 * DIM;
  bf16_t* wot  = wvt  + (size_t)512 * DIM;
  bf16_t* qb   = wot  + (size_t)DIM * DIM;
  bf16_t* kb   = qb   + (size_t)MTOT * 2048;
  bf16_t* vtb  = kb   + (size_t)MTOT * 512;
  bf16_t* attn = vtb  + (size_t)B_SZ * NKV * HD * SEQ;

  // 0) precision conversion (+ weight transpose to [N][K] for TDM tiling)
  cvt_bf16_kernel<<<(MTOT * DIM / 4) / 256, 256, 0, stream>>>(x, xb, MTOT * DIM / 4);
  cvtT_bf16_kernel<<<dim3(DIM / 32, DIM / 32), 256, 0, stream>>>(wq, wqt, DIM, DIM);
  cvtT_bf16_kernel<<<dim3(512 / 32, DIM / 32), 256, 0, stream>>>(wk, wkt, DIM, 512);
  cvtT_bf16_kernel<<<dim3(512 / 32, DIM / 32), 256, 0, stream>>>(wv, wvt, DIM, 512);
  cvtT_bf16_kernel<<<dim3(DIM / 32, DIM / 32), 256, 0, stream>>>(wo, wot, DIM, DIM);

  // 1) QKV projections (N = 2048 + 512 + 512 -> 24 column tiles)
  qkv_gemm_kernel<<<dim3(32, 24), dim3(256), 0, stream>>>(xb, wqt, wkt, wvt,
                                                          qb, kb, vtb);
  // 2) RoPE
  {
    int totq = MTOT * NH * 32;
    rope_kernel<<<totq / 256, 256, 0, stream>>>(qb, fc, fs, 5, totq);
    int totk = MTOT * NKV * 32;
    rope_kernel<<<totk / 256, 256, 0, stream>>>(kb, fc, fs, 3, totk);
  }
  // 3) Flash attention
  attn_kernel<<<dim3(SEQ / 64, NH, B_SZ), dim3(128), 0, stream>>>(qb, kb, vtb,
                                                                  mask, attn);
  // 4) Output projection
  out_gemm_kernel<<<dim3(32, 16), dim3(256), 0, stream>>>(attn, wot, out);
}
